// TensorProductLayer_34308198761128
// MI455X (gfx1250) — matrix operand order repaired and checked by reading, not compiled, not run on hardware
//
#include <hip/hip_runtime.h>
#include <hip/hip_bf16.h>
#include <math.h>

// ---------------------------------------------------------------------------
// e3nn FullyConnectedTensorProduct(512x0e+1x1o)^2 -> same, on MI455X (gfx1250)
//
// Dominant path: ss[t,w] = sum_{u,v} s[t,u] s[t,v] W[u,v,w]
//   = GEMM M=1024, N=512, K=262144 with A generated on the fly.
// f16 WMMA (16x16x32) with hi/lo error compensation (3 passes, ~2^-22 rel
// error ~= f32 quality): 8x the per-instruction MACs of f32 16x16x4 WMMA.
// W pre-converted once per call into blocked f16 hi/lo planes matching the
// dense B 32x16 fragment layout. Mt=32 token tile: every B fragment feeds
// 2 A-fragment pairs (24 WMMAs per 16 b128 loads) -> ~50 TB/s L2 demand at
// the f16 WMMA ceiling instead of ~100 TB/s with Mt=16. Split-K over u with
// deterministic partials + reduction (no float atomics).
// ---------------------------------------------------------------------------

typedef __attribute__((ext_vector_type(16))) _Float16 v16h;
typedef __attribute__((ext_vector_type(8)))  float    v8f;

#define T_TOK 1024          // B*N tokens
#define DD    512           // channel dim
#define KSPLIT 16           // u-chunks (split-K)
#define PLANE_ELEMS (512u * 512u * 512u)   // 134217728 f16 per plane

static __device__ __forceinline__ v16h load_v16h(const _Float16* p) {
  return *(const v16h*)p;   // 32B, 32B-aligned by construction
}

// Build one 16x32 f16 A-fragment pair (hi + residual-lo) for token rows
// srow[0..15], following the documented 16-bit A 16x32 VGPR layout:
// lane<16: K = e (e<8), 16+e-8 (e>=8); lane>=16: K = 8+e, 24+e-8.
// LDS rows are XOR-bank-swizzled by (tRow&15)<<2 (float4-granular).
static __device__ __forceinline__ void build_a_frags(
    const float* __restrict__ srow, const int sw, const float su,
    const int v0, const int kb, v16h* __restrict__ ah,
    v16h* __restrict__ al) {
  const float4 x0 = *(const float4*)(srow + (((v0 + kb) ^ sw)));
  const float4 x1 = *(const float4*)(srow + (((v0 + kb + 4) ^ sw)));
  const float4 x2 = *(const float4*)(srow + (((v0 + 16 + kb) ^ sw)));
  const float4 x3 = *(const float4*)(srow + (((v0 + 16 + kb + 4) ^ sw)));
  const float p[16] = {su * x0.x, su * x0.y, su * x0.z, su * x0.w,
                       su * x1.x, su * x1.y, su * x1.z, su * x1.w,
                       su * x2.x, su * x2.y, su * x2.z, su * x2.w,
                       su * x3.x, su * x3.y, su * x3.z, su * x3.w};
#pragma unroll
  for (int e = 0; e < 16; ++e) {
    const _Float16 h = (_Float16)p[e];
    (*ah)[e] = h;
    (*al)[e] = (_Float16)(p[e] - (float)h);
  }
}

// ---------------------------------------------------------------------------
// Kernel 1: convert W_sss (f32 [512][512][512], w fastest) into blocked f16
// hi/lo planes. Tile (u, vb, wb): 32 v x 16 w, stored in the exact per-lane
// order of the dense 16-bit B 32x16 WMMA fragment: lane L holds column
// w = wb*16 + (L&15), K v = vb*32 + (L>>4)*8 + (e<8 ? e : e+8) -> 32B contiguous.
// ---------------------------------------------------------------------------
__global__ __launch_bounds__(256) void prep_w_kernel(
    const float* __restrict__ W, _Float16* __restrict__ whi,
    _Float16* __restrict__ wlo) {
  const int tid    = threadIdx.x;
  const int tileId = blockIdx.x * 8 + (tid >> 5);   // 262144 tiles total
  const int lane   = tid & 31;
  const int u  = tileId >> 9;          // 0..511
  const int vb = (tileId >> 5) & 15;   // 0..15
  const int wb = tileId & 31;          // 0..31
  const int w  = wb * 16 + (lane & 15);
  const int v0 = vb * 32 + (lane >> 4) * 8;

  v16h hv, lv;
#pragma unroll
  for (int e = 0; e < 16; ++e) {
    const float x = W[((size_t)(u * 512 + (v0 + (e < 8 ? e : e + 8))) << 9) + w];
    const _Float16 h = (_Float16)x;
    const float r = x - (float)h;
    hv[e] = h;
    lv[e] = (_Float16)r;
  }
  const size_t off = (size_t)tileId * 512 + (size_t)lane * 16;
  *(v16h*)(whi + off) = hv;
  *(v16h*)(wlo + off) = lv;
}

// ---------------------------------------------------------------------------
// Kernel 2: main GEMM. grid = (32 token-tiles, KSPLIT u-chunks), block = 256
// (8 waves). Wave r owns w columns [r*64, r*64+64) for 32 tokens:
// 8 accumulator tiles; each A build feeds 24 WMMAs, each B load feeds 6.
// ---------------------------------------------------------------------------
__global__ __launch_bounds__(256) void gemm_f16_split(
    const float* __restrict__ s,          // [1024][512]
    const _Float16* __restrict__ whi,     // blocked plane
    const _Float16* __restrict__ wlo,     // blocked plane
    float* __restrict__ partials) {       // [KSPLIT][1024][512]
  __shared__ float sld[32 * 512];         // 64 KB, XOR-bank-swizzled rows

  const int tid = threadIdx.x;
  const int t0  = blockIdx.x * 32;
  const int uc  = blockIdx.y;             // 0..KSPLIT-1
  const int u0  = uc * (512 / KSPLIT);

  // Stage s tile (32 tokens x 512) into LDS with float4-granular XOR swizzle.
  for (int i = tid; i < 32 * 128; i += 256) {
    const int t  = i >> 7;
    const int v4 = (i & 127) << 2;
    const int vs = v4 ^ ((t & 15) << 2);
    const float4 f = *(const float4*)(s + (size_t)(t0 + t) * DD + v4);
    float* d = &sld[t * 512 + vs];
    d[0] = f.x; d[1] = f.y; d[2] = f.z; d[3] = f.w;
  }
  __syncthreads();

  const int lane = tid & 31;
  const int wave = tid >> 5;
  const int tRow = lane & 15;             // A-matrix M row for this lane
  const int kb   = (lane >> 4) * 8;       // K sub-base per 16-bit A layout
  const int sw   = (tRow & 15) << 2;      // bank swizzle for this row
  const float* srow0 = &sld[tRow * 512];
  const float* srow1 = &sld[(16 + tRow) * 512];

  v8f c0[4] = {}, c1[4] = {};             // 8 16x16 f32 accumulators

  for (int u = u0; u < u0 + (512 / KSPLIT); ++u) {
    const float su0 = srow0[u ^ sw];
    const float su1 = srow1[u ^ sw];
    for (int vb = 0; vb < 16; ++vb) {
      const int v0 = vb * 32;
      v16h ah0, al0, ah1, al1;
      build_a_frags(srow0, sw, su0, v0, kb, &ah0, &al0);
      build_a_frags(srow1, sw, su1, v0, kb, &ah1, &al1);

      const size_t tbase =
          ((size_t)((u * 16 + vb) * 32 + wave * 4)) * 512 + (size_t)lane * 16;
      // Prefetch next v-block's B tiles (global_prefetch_b8).
      __builtin_prefetch(whi + tbase + 32 * 512, 0, 1);
      __builtin_prefetch(wlo + tbase + 32 * 512, 0, 1);

#pragma unroll
      for (int ws = 0; ws < 4; ++ws) {
        const v16h bh = load_v16h(whi + tbase + (size_t)ws * 512);
        const v16h bl = load_v16h(wlo + tbase + (size_t)ws * 512);
        // 3 passes per token-half: Ah*Bh + Al*Bh + Ah*Bl (~2^-22 rel. error)
        c0[ws] = __builtin_amdgcn_wmma_f32_16x16x32_f16(
            false, ah0, false, bh, (short)0, c0[ws], false, false);
        c0[ws] = __builtin_amdgcn_wmma_f32_16x16x32_f16(
            false, al0, false, bh, (short)0, c0[ws], false, false);
        c0[ws] = __builtin_amdgcn_wmma_f32_16x16x32_f16(
            false, ah0, false, bl, (short)0, c0[ws], false, false);
        c1[ws] = __builtin_amdgcn_wmma_f32_16x16x32_f16(
            false, ah1, false, bh, (short)0, c1[ws], false, false);
        c1[ws] = __builtin_amdgcn_wmma_f32_16x16x32_f16(
            false, al1, false, bh, (short)0, c1[ws], false, false);
        c1[ws] = __builtin_amdgcn_wmma_f32_16x16x32_f16(
            false, ah1, false, bl, (short)0, c1[ws], false, false);
      }
    }
  }

  // Epilogue: C/D 16x16 layout -> partials[uc][t][w].
  const int colW = lane & 15;
  const int rOff = (lane >> 4) * 8;
#pragma unroll
  for (int ws = 0; ws < 4; ++ws) {
    const int wcol = wave * 64 + ws * 16 + colW;
#pragma unroll
    for (int r = 0; r < 8; ++r) {
      partials[((size_t)(uc * T_TOK + (t0 + r + rOff)) << 9) + wcol] =
          c0[ws][r];
      partials[((size_t)(uc * T_TOK + (t0 + 16 + r + rOff)) << 9) + wcol] =
          c1[ws][r];
    }
  }
}

// ---------------------------------------------------------------------------
// Fallback GEMM (VALU only) if scratch is too small for the f16 planes.
// grid = (32 t-tiles, 4 w-tiles, KSPLIT), block = 256.
// ---------------------------------------------------------------------------
__global__ __launch_bounds__(256) void gemm_fallback(
    const float* __restrict__ s, const float* __restrict__ W,
    float* __restrict__ partials) {
  __shared__ float sld[32 * 512];         // 64 KB
  const int tid = threadIdx.x;
  const int t0 = blockIdx.x * 32;
  const int w0 = blockIdx.y * 128;
  const int uc = blockIdx.z;
  for (int i = tid; i < 32 * 128; i += 256) {
    const int t = i >> 7;
    const int v4 = (i & 127) << 2;
    *(float4*)&sld[t * 512 + v4] =
        *(const float4*)(s + (size_t)(t0 + t) * DD + v4);
  }
  __syncthreads();
  const int w = w0 + (tid & 127);
  const int tHalf = tid >> 7;             // 0/1 -> 16 tokens each
  float acc[16] = {};
  for (int u = uc * (512 / KSPLIT); u < (uc + 1) * (512 / KSPLIT); ++u) {
    float su[16];
#pragma unroll
    for (int i = 0; i < 16; ++i) su[i] = sld[(tHalf * 16 + i) * 512 + u];
    for (int v = 0; v < 512; ++v) {
      const float wv = W[((size_t)(u * 512 + v) << 9) + w];
#pragma unroll
      for (int i = 0; i < 16; ++i)
        acc[i] += su[i] * sld[(tHalf * 16 + i) * 512 + v] * wv;
    }
  }
#pragma unroll
  for (int i = 0; i < 16; ++i) {
    const int t = t0 + tHalf * 16 + i;
    partials[((size_t)(uc * T_TOK + t) << 9) + w] = acc[i];
  }
}

// ---------------------------------------------------------------------------
// Kernel 3: reduce split-K partials + vv path -> scalar_out (deterministic).
// ---------------------------------------------------------------------------
__global__ __launch_bounds__(256) void reduce_scalar_kernel(
    const float* __restrict__ partials, const float* __restrict__ vectors,
    const float* __restrict__ W_vvs, float* __restrict__ out) {
  const int t = blockIdx.x;
  const float inv_sqrt3 = 0.57735026918962576f;
  const float alpha_s = 1.0f / sqrtf(512.0f * 512.0f + 1.0f);
  const float vx = vectors[t * 3 + 0];
  const float vy = vectors[t * 3 + 1];
  const float vz = vectors[t * 3 + 2];
  const float vv = inv_sqrt3 * (vx * vx + vy * vy + vz * vz);
  for (int w = threadIdx.x; w < DD; w += 256) {
    float acc = 0.0f;
#pragma unroll
    for (int k = 0; k < KSPLIT; ++k)
      acc += partials[((size_t)(k * T_TOK + t) << 9) + w];
    out[3072 + (size_t)t * DD + w] = alpha_s * (acc + vv * W_vvs[w]);
  }
}

// ---------------------------------------------------------------------------
// Kernel 4: vector path. One wave32 per token; alpha_v/sqrt(3) == 1/32.
// ---------------------------------------------------------------------------
__global__ __launch_bounds__(256) void vector_out_kernel(
    const float* __restrict__ s, const float* __restrict__ vectors,
    const float* __restrict__ W_svv, const float* __restrict__ W_vsv,
    float* __restrict__ out) {
  const int lane = threadIdx.x & 31;
  const int wv = threadIdx.x >> 5;
  const int t = blockIdx.x * 8 + wv;
  float acc = 0.0f;
  for (int i = lane; i < DD; i += 32)
    acc += s[(size_t)t * DD + i] * (W_svv[i] + W_vsv[i]);
#pragma unroll
  for (int off = 16; off > 0; off >>= 1) acc += __shfl_xor(acc, off, 32);
  const float coef = acc * 0.03125f;      // sqrt(3/(2*512)) / sqrt(3) = 1/32
  if (lane < 3) out[t * 3 + lane] = coef * vectors[t * 3 + lane];
}

// ---------------------------------------------------------------------------
extern "C" void kernel_launch(void* const* d_in, const int* in_sizes, int n_in,
                              void* d_out, int out_size, void* d_ws,
                              size_t ws_size, hipStream_t stream) {
  const float* vectors = (const float*)d_in[0];   // [2,512,3]
  const float* scalars = (const float*)d_in[1];   // [2,512,512]
  const float* W_sss   = (const float*)d_in[2];   // [512,512,512]
  const float* W_vvs   = (const float*)d_in[3];
  const float* W_svv   = (const float*)d_in[4];
  const float* W_vsv   = (const float*)d_in[5];
  float* out = (float*)d_out;                     // [vec 3072 | scal 524288]

  const size_t PART_BYTES = (size_t)KSPLIT * T_TOK * DD * sizeof(float); // 32MB
  const size_t PLANE_BYTES = (size_t)PLANE_ELEMS * sizeof(_Float16);     // 256MB
  float* partials = (float*)d_ws;
  _Float16* whi = (_Float16*)((char*)d_ws + PART_BYTES);
  _Float16* wlo = whi + PLANE_ELEMS;

  const bool fast = ws_size >= PART_BYTES + 2 * PLANE_BYTES;
  if (fast) {
    prep_w_kernel<<<32768, 256, 0, stream>>>(W_sss, whi, wlo);
    gemm_f16_split<<<dim3(T_TOK / 32, KSPLIT), 256, 0, stream>>>(
        scalars, whi, wlo, partials);
  } else {
    gemm_fallback<<<dim3(T_TOK / 32, 4, KSPLIT), 256, 0, stream>>>(
        scalars, W_sss, partials);
  }
  reduce_scalar_kernel<<<T_TOK, 256, 0, stream>>>(partials, vectors, W_vvs, out);
  vector_out_kernel<<<T_TOK / 8, 256, 0, stream>>>(scalars, vectors, W_svv,
                                                   W_vsv, out);
}
